// MultiHeadedAttention_28836410425749
// MI455X (gfx1250) — compile-verified, matrix operand order checked by reading, not run on hardware
//
#include <hip/hip_runtime.h>
#include <hip/hip_bf16.h>

typedef __attribute__((ext_vector_type(16))) __bf16 v16bf;
typedef __attribute__((ext_vector_type(8)))  __bf16 v8bf;
typedef __attribute__((ext_vector_type(8)))  float  v8f;

#define B_  2
#define S_  2048
#define D_  1024
#define H_  16
#define HD_ 64
#define M_  (B_ * S_)   // 4096

// ---------------------------------------------------------------------------
// CDNA5 async global->LDS copy (GLOBAL_LOAD_ASYNC_TO_LDS_B128, ASYNCcnt).
// LDS operand is the low 32 bits of the generic pointer (flat->LDS truncation
// per ISA 10.2); global operand is a 64-bit VGPR pair. Inline asm is the
// toolchain-portable route (bypasses clang builtin arity differences).
// ---------------------------------------------------------------------------
__device__ __forceinline__ void async_b128(void* lds, const void* g) {
    unsigned l = (unsigned)(size_t)lds;
    asm volatile("global_load_async_to_lds_b128 %0, %1, off"
                 :: "v"(l), "v"(g) : "memory");
}
#define WAIT_ASYNC(n) asm volatile("s_wait_asynccnt " #n ::: "memory")

// ---------------------------------------------------------------------------
// Fragment assembly: CDNA5 16-bit A/B layout. Lane L holds (row/col = L%16),
// with 16 K-values as two contiguous 8-element runs at offsets
// (half*8) and (16 + half*8) within each 32-wide K step, half = L/16.
// Each run of 8 bf16 = 16 bytes = one b128 load.
// ---------------------------------------------------------------------------
__device__ __forceinline__ v16bf frag2(const __bf16* p0, const __bf16* p1) {
    v8bf lo = *(const v8bf*)p0;
    v8bf hi = *(const v8bf*)p1;
    return __builtin_shufflevector(lo, hi,
        0, 1, 2, 3, 4, 5, 6, 7, 8, 9, 10, 11, 12, 13, 14, 15);
}

__device__ __forceinline__ v8f wmma_bf16(v16bf a, v16bf b, v8f c) {
    return __builtin_amdgcn_wmma_f32_16x16x32_bf16(
        false, a, false, b, (short)0, c, false, false);
}

__device__ __forceinline__ float redMax16(float v) {
    v = fmaxf(v, __shfl_xor(v, 1, 32));
    v = fmaxf(v, __shfl_xor(v, 2, 32));
    v = fmaxf(v, __shfl_xor(v, 4, 32));
    v = fmaxf(v, __shfl_xor(v, 8, 32));
    return v;
}
__device__ __forceinline__ float redSum16(float v) {
    v += __shfl_xor(v, 1, 32);
    v += __shfl_xor(v, 2, 32);
    v += __shfl_xor(v, 4, 32);
    v += __shfl_xor(v, 8, 32);
    return v;
}

// ---------------------------------------------------------------------------
// fp32 -> bf16 elementwise convert
// ---------------------------------------------------------------------------
__global__ __launch_bounds__(256)
void cvt_bf16_kernel(const float* __restrict__ in,
                     __bf16* __restrict__ out, int n) {
    int i = (blockIdx.x * blockDim.x + threadIdx.x) * 4;
    if (i + 3 < n) {
        float4 f = *(const float4*)(in + i);
        out[i + 0] = (__bf16)f.x;
        out[i + 1] = (__bf16)f.y;
        out[i + 2] = (__bf16)f.z;
        out[i + 3] = (__bf16)f.w;
    }
}

// ---------------------------------------------------------------------------
// W[K][N] fp32  ->  Wt[N][K] bf16  (32x32 LDS tile transpose)
// block (32, 8); grid (N/32, K/32)
// ---------------------------------------------------------------------------
__global__ __launch_bounds__(256)
void transpose_cvt_kernel(const float* __restrict__ W,
                          __bf16* __restrict__ Wt) {
    __shared__ float tile[32][33];
    const int nBase = blockIdx.x * 32;
    const int kBase = blockIdx.y * 32;
    const int tx = threadIdx.x, ty = threadIdx.y;
#pragma unroll
    for (int i = 0; i < 4; ++i)
        tile[ty + 8 * i][tx] = W[(size_t)(kBase + ty + 8 * i) * D_ + nBase + tx];
    __syncthreads();
#pragma unroll
    for (int i = 0; i < 4; ++i)
        Wt[(size_t)(nBase + ty + 8 * i) * D_ + kBase + tx] =
            (__bf16)tile[tx][ty + 8 * i];
}

// ---------------------------------------------------------------------------
// GEMM: C[M,N] = A[M,K] * Wt[N,K]^T + bias[N]
// A, Wt bf16; accum fp32.  M=4096, N=K=1024.
// Block: 256 threads (8 waves, 2x4), tile 128x128, K-step 32.
// Async-to-LDS double buffer: tile kt+1 streams via ASYNCcnt while tile kt
// is consumed by WMMA; s_wait_asynccnt <= 4 retires exactly the prior tile.
// MODE 0: store bf16 into [b,h,s,hd]    (q, k projections)
// MODE 2: store bf16 into [b,h,hd,s]    (v projection, transposed)
// MODE 3: store fp32 row-major [M,N]    (final output)
// ---------------------------------------------------------------------------
template <int MODE>
__global__ __launch_bounds__(256) __attribute__((amdgpu_waves_per_eu(2)))
void gemm_bf16_kernel(const __bf16* __restrict__ A,
                      const __bf16* __restrict__ Bt,
                      const float* __restrict__ bias,
                      void* __restrict__ outv) {
    __shared__ __align__(16) __bf16 aT[2][128 * 32];
    __shared__ __align__(16) __bf16 bT[2][128 * 32];

    const int tid  = threadIdx.x;
    const int lane = tid & 31;
    const int wid  = tid >> 5;
    const int half = lane >> 4;
    const int l16  = lane & 15;
    const int waveM = wid >> 2;   // 0..1
    const int waveN = wid & 3;    // 0..3
    const int mBase = blockIdx.y * 128;
    const int nBase = blockIdx.x * 128;

    // each thread owns 2 b128 chunks per tile (512 chunks / 256 threads)
    const int c0r = tid >> 2;            // chunk row for i=0
    const int c0p = (tid & 3) * 8;       // chunk col (bf16 elems)
    const int c1r = (tid + 256) >> 2;    // chunk row for i=1
    const int c1p = c0p;

    const __bf16* aSrc0 = A  + (size_t)(mBase + c0r) * D_ + c0p;
    const __bf16* aSrc1 = A  + (size_t)(mBase + c1r) * D_ + c1p;
    const __bf16* bSrc0 = Bt + (size_t)(nBase + c0r) * D_ + c0p;
    const __bf16* bSrc1 = Bt + (size_t)(nBase + c1r) * D_ + c1p;

    auto issueTile = [&](int kt, int buf) {
        const int kOff = kt * 32;
        async_b128(aT[buf] + c0r * 32 + c0p, aSrc0 + kOff);
        async_b128(aT[buf] + c1r * 32 + c1p, aSrc1 + kOff);
        async_b128(bT[buf] + c0r * 32 + c0p, bSrc0 + kOff);
        async_b128(bT[buf] + c1r * 32 + c1p, bSrc1 + kOff);
    };

    v8f acc[4][2];
#pragma unroll
    for (int mt = 0; mt < 4; ++mt)
#pragma unroll
        for (int nt = 0; nt < 2; ++nt)
            acc[mt][nt] = (v8f){0, 0, 0, 0, 0, 0, 0, 0};

    issueTile(0, 0);

    const int NKT = D_ / 32;  // 32
    for (int kt = 0; kt < NKT; ++kt) {
        const int cur = kt & 1;
        const bool more = (kt + 1) < NKT;
        if (more) {
            issueTile(kt + 1, cur ^ 1);
            WAIT_ASYNC(4);   // retire tile kt (in-order), keep kt+1 in flight
        } else {
            WAIT_ASYNC(0);
        }
        __syncthreads();

        v16bf aF[4], bF[2];
#pragma unroll
        for (int mt = 0; mt < 4; ++mt) {
            const __bf16* rp = aT[cur] + (waveM * 64 + mt * 16 + l16) * 32;
            aF[mt] = frag2(rp + half * 8, rp + 16 + half * 8);
        }
#pragma unroll
        for (int nt = 0; nt < 2; ++nt) {
            const __bf16* rp = bT[cur] + (waveN * 32 + nt * 16 + l16) * 32;
            bF[nt] = frag2(rp + half * 8, rp + 16 + half * 8);
        }
#pragma unroll
        for (int mt = 0; mt < 4; ++mt)
#pragma unroll
            for (int nt = 0; nt < 2; ++nt)
                acc[mt][nt] = wmma_bf16(aF[mt], bF[nt], acc[mt][nt]);

        __syncthreads();   // all waves done reading buf cur before overwrite
    }

    // epilogue: lane holds col n = l16, rows r + 8*half per tile
#pragma unroll
    for (int mt = 0; mt < 4; ++mt) {
#pragma unroll
        for (int nt = 0; nt < 2; ++nt) {
            const int nG = nBase + waveN * 32 + nt * 16 + l16;
            const float bv = bias[nG];
#pragma unroll
            for (int r = 0; r < 8; ++r) {
                const int mG = mBase + waveM * 64 + mt * 16 + r + 8 * half;
                const float val = acc[mt][nt][r] + bv;
                if (MODE == 3) {
                    ((float*)outv)[(size_t)mG * D_ + nG] = val;
                } else {
                    const int b  = mG >> 11;       // / S_
                    const int s  = mG & (S_ - 1);
                    const int h  = nG >> 6;        // / HD_
                    const int hd = nG & (HD_ - 1);
                    __bf16* o = (__bf16*)outv;
                    if (MODE == 2)
                        o[((size_t)(b * H_ + h) * HD_ + hd) * S_ + s] = (__bf16)val;
                    else
                        o[((size_t)(b * H_ + h) * S_ + s) * HD_ + hd] = (__bf16)val;
                }
            }
        }
    }
}

// ---------------------------------------------------------------------------
// Flash attention: grid (S/128, B*H), block 256 (8 waves).
// Wave w owns Q rows [qBase + 16w, +16), Q fragments register-resident.
// K / Vt blocks double-buffered in LDS and streamed with async-to-LDS:
// block kb+1 transfers while block kb is computed (QK^T -> online softmax
// -> P V, all WMMA). 8 async ops/thread/block; s_wait_asynccnt <= 8 retires
// exactly the prior block.
// ---------------------------------------------------------------------------
__global__ __launch_bounds__(256) __attribute__((amdgpu_waves_per_eu(2)))
void attn_kernel(const __bf16* __restrict__ qp,
                 const __bf16* __restrict__ kp,
                 const __bf16* __restrict__ vtp,
                 const int* __restrict__ mask,
                 __bf16* __restrict__ attnOut) {
    __shared__ __align__(16) __bf16 kT[2][128 * 64];     // 2 x 16 KB
    __shared__ __align__(16) __bf16 vT[2][64 * 128];     // 2 x 16 KB
    __shared__ __align__(16) __bf16 pT[8 * 16 * 128];    // 32 KB

    const int tid  = threadIdx.x;
    const int lane = tid & 31;
    const int wid  = tid >> 5;
    const int half = lane >> 4;
    const int l16  = lane & 15;
    const int bh   = blockIdx.y;
    const int b    = bh >> 4;
    const int h    = bh & 15;
    const int qBase = blockIdx.x * 128;

    const __bf16* kSrc = kp + (size_t)bh * S_ * HD_;
    const __bf16* vSrc = vtp + (size_t)bh * HD_ * S_;

    auto issueBlock = [&](int kb, int buf) {
        const int s0 = kb * 128;
        // K block: rows [s0, s0+128) x 64 hd — contiguous 16 KB
#pragma unroll
        for (int i = 0; i < 4; ++i) {
            const int c = tid + i * 256;          // 1024 b128 chunks
            async_b128(kT[buf] + c * 8, kSrc + (size_t)s0 * HD_ + c * 8);
        }
        // Vt block: 64 hd-rows x 128 s, 256 B per row from stride-S rows
#pragma unroll
        for (int i = 0; i < 4; ++i) {
            const int c  = tid + i * 256;
            const int hd = c >> 4;
            const int p  = (c & 15) * 8;
            async_b128(vT[buf] + hd * 128 + p,
                       vSrc + (size_t)hd * S_ + s0 + p);
        }
    };

    // Q fragments for this wave's 16 rows, held in registers for all blocks
    const __bf16* qrow = qp + ((size_t)bh * S_ + qBase + wid * 16 + l16) * HD_;
    v16bf aQ[2];
    aQ[0] = frag2(qrow + half * 8, qrow + 16 + half * 8);
    aQ[1] = frag2(qrow + 32 + half * 8, qrow + 48 + half * 8);

    float mrun[8], lrun[8];
    v8f oAcc[4];
#pragma unroll
    for (int r = 0; r < 8; ++r) { mrun[r] = -1e30f; lrun[r] = 0.f; }
#pragma unroll
    for (int ot = 0; ot < 4; ++ot) oAcc[ot] = (v8f){0, 0, 0, 0, 0, 0, 0, 0};

    issueBlock(0, 0);

    const int NB = S_ / 128;  // 16
    for (int kb = 0; kb < NB; ++kb) {
        const int cur = kb & 1;
        const int s0 = kb * 128;
        const bool more = (kb + 1) < NB;
        if (more) {
            issueBlock(kb + 1, cur ^ 1);
            WAIT_ASYNC(8);   // retire block kb (in-order), keep kb+1 in flight
        } else {
            WAIT_ASYNC(0);
        }
        __syncthreads();

        // ---- scores = Q Kt ----
        v8f sc[8];
#pragma unroll
        for (int nt = 0; nt < 8; ++nt) sc[nt] = (v8f){0, 0, 0, 0, 0, 0, 0, 0};
#pragma unroll
        for (int nt = 0; nt < 8; ++nt) {
            const __bf16* kr = kT[cur] + (nt * 16 + l16) * 64;
            v16bf b0 = frag2(kr + half * 8, kr + 16 + half * 8);
            v16bf b1 = frag2(kr + 32 + half * 8, kr + 48 + half * 8);
            sc[nt] = wmma_bf16(aQ[0], b0, sc[nt]);
            sc[nt] = wmma_bf16(aQ[1], b1, sc[nt]);
        }

        // ---- scale + mask + online softmax ----
        float rmax[8];
#pragma unroll
        for (int r = 0; r < 8; ++r) rmax[r] = -1e30f;
#pragma unroll
        for (int nt = 0; nt < 8; ++nt) {
            const int mv = mask[b * S_ + s0 + nt * 16 + l16];
#pragma unroll
            for (int r = 0; r < 8; ++r) {
                float v = sc[nt][r] * 0.125f;
                v = mv ? v : -1e10f;
                sc[nt][r] = v;
                rmax[r] = fmaxf(rmax[r], v);
            }
        }
        float nm[8], fac[8], rs[8];
#pragma unroll
        for (int r = 0; r < 8; ++r) {
            rmax[r] = redMax16(rmax[r]);
            nm[r]   = fmaxf(mrun[r], rmax[r]);
            fac[r]  = __expf(mrun[r] - nm[r]);
            mrun[r] = nm[r];
            rs[r]   = 0.f;
        }
#pragma unroll
        for (int nt = 0; nt < 8; ++nt)
#pragma unroll
            for (int r = 0; r < 8; ++r) {
                float p = __expf(sc[nt][r] - nm[r]);
                sc[nt][r] = p;
                rs[r] += p;
            }
#pragma unroll
        for (int r = 0; r < 8; ++r) {
            rs[r]   = redSum16(rs[r]);
            lrun[r] = lrun[r] * fac[r] + rs[r];
        }
#pragma unroll
        for (int ot = 0; ot < 4; ++ot)
#pragma unroll
            for (int r = 0; r < 8; ++r) oAcc[ot][r] *= fac[r];

        // ---- P -> LDS (bf16); per-wave region, same-wave DS ordering
        //      (compiler inserts s_wait_dscnt), so no barrier needed ----
        __bf16* pw = pT + wid * 16 * 128;
#pragma unroll
        for (int nt = 0; nt < 8; ++nt)
#pragma unroll
            for (int r = 0; r < 8; ++r)
                pw[(r + 8 * half) * 128 + nt * 16 + l16] = (__bf16)sc[nt][r];

        // ---- O += P V ----
        const __bf16* pr = pT + wid * 16 * 128 + l16 * 128;
#pragma unroll
        for (int ot = 0; ot < 4; ++ot) {
            const __bf16* vr = vT[cur] + (ot * 16 + l16) * 128;
#pragma unroll
            for (int ks = 0; ks < 4; ++ks) {
                v16bf aP = frag2(pr + ks * 32 + half * 8,
                                 pr + ks * 32 + 16 + half * 8);
                v16bf bV = frag2(vr + ks * 32 + half * 8,
                                 vr + ks * 32 + 16 + half * 8);
                oAcc[ot] = wmma_bf16(aP, bV, oAcc[ot]);
            }
        }
        __syncthreads();   // all waves done with buf cur before overwrite
    }

    // ---- epilogue: normalize and write bf16 [b*s, D] ----
#pragma unroll
    for (int ot = 0; ot < 4; ++ot)
#pragma unroll
        for (int r = 0; r < 8; ++r) {
            const int sRow = qBase + wid * 16 + r + 8 * half;
            const float val = oAcc[ot][r] / lrun[r];
            attnOut[(size_t)(b * S_ + sRow) * D_ + h * HD_ + ot * 16 + l16] =
                (__bf16)val;
        }
}

// ---------------------------------------------------------------------------
// Host launcher
// ---------------------------------------------------------------------------
extern "C" void kernel_launch(void* const* d_in, const int* in_sizes, int n_in,
                              void* d_out, int out_size, void* d_ws,
                              size_t ws_size, hipStream_t stream) {
    const float* query = (const float*)d_in[0];
    const float* key   = (const float*)d_in[1];
    const float* value = (const float*)d_in[2];
    const int*   mask  = (const int*)d_in[3];
    const float* Wq = (const float*)d_in[4];
    const float* bq = (const float*)d_in[5];
    const float* Wk = (const float*)d_in[6];
    const float* bk = (const float*)d_in[7];
    const float* Wv = (const float*)d_in[8];
    const float* bv = (const float*)d_in[9];
    const float* Wo = (const float*)d_in[10];
    const float* bo = (const float*)d_in[11];
    float* out = (float*)d_out;

    const size_t MK = (size_t)M_ * D_;  // 4M elems
    const size_t KK = (size_t)D_ * D_;  // 1M elems
    __bf16* ws = (__bf16*)d_ws;
    __bf16* qin = ws;            ws += MK;
    __bf16* kin = ws;            ws += MK;
    __bf16* vin = ws;            ws += MK;
    __bf16* WqT = ws;            ws += KK;
    __bf16* WkT = ws;            ws += KK;
    __bf16* WvT = ws;            ws += KK;
    __bf16* WoT = ws;            ws += KK;
    __bf16* qpb = ws;            ws += MK;   // [b,h,s,hd]
    __bf16* kpb = ws;            ws += MK;   // [b,h,s,hd]
    __bf16* vtb = ws;            ws += MK;   // [b,h,hd,s]
    __bf16* atn = ws;            ws += MK;   // [b*s, D]

    const dim3 cb(256);
    cvt_bf16_kernel<<<dim3(MK / 1024), cb, 0, stream>>>(query, qin, (int)MK);
    cvt_bf16_kernel<<<dim3(MK / 1024), cb, 0, stream>>>(key,   kin, (int)MK);
    cvt_bf16_kernel<<<dim3(MK / 1024), cb, 0, stream>>>(value, vin, (int)MK);

    const dim3 tb(32, 8), tg(D_ / 32, D_ / 32);
    transpose_cvt_kernel<<<tg, tb, 0, stream>>>(Wq, WqT);
    transpose_cvt_kernel<<<tg, tb, 0, stream>>>(Wk, WkT);
    transpose_cvt_kernel<<<tg, tb, 0, stream>>>(Wv, WvT);
    transpose_cvt_kernel<<<tg, tb, 0, stream>>>(Wo, WoT);

    const dim3 gb(256);
    const dim3 gg(D_ / 128, M_ / 128);  // (8, 32)
    gemm_bf16_kernel<0><<<gg, gb, 0, stream>>>(qin, WqT, bq, qpb);
    gemm_bf16_kernel<0><<<gg, gb, 0, stream>>>(kin, WkT, bk, kpb);
    gemm_bf16_kernel<2><<<gg, gb, 0, stream>>>(vin, WvT, bv, vtb);

    const dim3 ag(S_ / 128, B_ * H_);   // (16, 32)
    attn_kernel<<<ag, gb, 0, stream>>>(qpb, kpb, vtb, mask, atn);

    gemm_bf16_kernel<3><<<gg, gb, 0, stream>>>(atn, WoT, bo, (void*)out);
}
